// GeneralConv_11596411700025
// MI455X (gfx1250) — matrix operand order repaired and checked by reading, not compiled, not run on hardware
//
#include <hip/hip_runtime.h>
#include <hip/hip_bf16.h>

#define D   128
#define H   8
#define DKK 16

typedef __attribute__((ext_vector_type(16))) __bf16 v16bf;
typedef __attribute__((ext_vector_type(8)))  float  v8f;

__device__ __forceinline__ unsigned bfbits(float f) {   // bf16 bits, RNE
  unsigned u = __float_as_uint(f);
  return (u + 0x7FFFu + ((u >> 16) & 1u)) >> 16;
}
__device__ __forceinline__ __bf16 f2bf(float f) {
  unsigned short h = (unsigned short)bfbits(f);
  return __builtin_bit_cast(__bf16, h);
}
__device__ __forceinline__ float bf2f(__bf16 b) {
  unsigned short h = __builtin_bit_cast(unsigned short, b);
  return __uint_as_float(((unsigned)h) << 16);
}

// ---- WMMA fragment loaders (ISA 7.12.2 layouts, wave32) -------------------
// A: 16x32 bf16 tile from row-major LDS [16][128].
// lanes 0-15: M=lane, K = {k0..k0+7, k0+16..k0+23}; lanes 16-31: K shifted +8.
__device__ __forceinline__ v16bf load_A_frag(const __bf16* As, int lane, int k0) {
  int m  = lane & 15;
  int kb = k0 + ((lane & 16) ? 8 : 0);
  const __bf16* row = As + m * D + kb;
  v16bf a;
#pragma unroll
  for (int i = 0; i < 8; ++i) a[i] = row[i];
#pragma unroll
  for (int i = 0; i < 8; ++i) a[8 + i] = row[16 + i];
  return a;
}
// B fragment (32x16) built directly from the global fp32 weight tensor
// W[H][D][16]; columns of head `head`. Loaded ONCE per wave, lives in VGPRs.
__device__ __forceinline__ v16bf load_B_frag_global(const float* __restrict__ W,
                                                    int lane, int k0, int head) {
  int kk = lane & 15;
  int kb = k0 + ((lane & 16) ? 8 : 0);
  v16bf b;
#pragma unroll
  for (int i = 0; i < 16; ++i)
    b[i] = f2bf(W[((size_t)head * D + kb + i + (i >= 8 ? 8 : 0)) * DKK + kk]);
  return b;
}
__device__ __forceinline__ v8f wmma_bf16(v16bf a, v16bf b, v8f c) {
  return __builtin_amdgcn_wmma_f32_16x16x32_bf16(false, a, false, b, (short)0, c, false, false);
}

// ---- kernels ---------------------------------------------------------------

__global__ void k_init(float* out, float* deg, unsigned* maxu, float* den, int N) {
  int i = blockIdx.x * blockDim.x + threadIdx.x;
  if (i < N * D) out[i] = 0.f;
  if (i < N)      deg[i] = 0.f;
  if (i < N * H) { maxu[i] = 0u; den[i] = 0.f; }
}

// one wave per node row; float4 per lane (coalesced b128)
__global__ void k_layernorm(const float* __restrict__ x, float* __restrict__ xn, int N) {
  int row  = (blockIdx.x * blockDim.x + threadIdx.x) >> 5;
  int lane = threadIdx.x & 31;
  if (row >= N) return;
  float4 v = ((const float4*)(x + (size_t)row * D))[lane];
  float s = v.x + v.y + v.z + v.w;
#pragma unroll
  for (int m = 16; m >= 1; m >>= 1) s += __shfl_xor(s, m, 32);
  float mu = s * (1.f / 128.f);
  float dx = v.x - mu, dy = v.y - mu, dz = v.z - mu, dw = v.w - mu;
  float vs = dx * dx + dy * dy + dz * dz + dw * dw;
#pragma unroll
  for (int m = 16; m >= 1; m >>= 1) vs += __shfl_xor(vs, m, 32);
  float inv = rsqrtf(vs * (1.f / 128.f) + 1e-5f);
  float4 o; o.x = dx * inv; o.y = dy * inv; o.z = dz * inv; o.w = dw * inv;
  ((float4*)(xn + (size_t)row * D))[lane] = o;
}

__global__ void k_deg(const long long* __restrict__ ei, const float* __restrict__ w,
                      float* __restrict__ deg, int E) {
  int e = blockIdx.x * blockDim.x + threadIdx.x;
  if (e < E) atomicAdd(&deg[ei[e]], w[e]);
}

// q_node = LN(x) @ Wq + bq   (node-level GEMM, WMMA; B held in registers)
__global__ void k_proj_q(const float* __restrict__ xn, const float* __restrict__ Wq,
                         const float* __restrict__ bq, float* __restrict__ q, int N) {
  __shared__ __bf16 As[16 * D];
  int tid = threadIdx.x, lane = tid & 31, w = tid >> 5;   // 8 waves, wave = head/n-tile
  v16bf bf[4];
#pragma unroll
  for (int ks = 0; ks < 4; ++ks) bf[ks] = load_B_frag_global(Wq, lane, ks * 32, w);
  int n = w * DKK + (lane & 15);
  float bias = bq[n];
  int ntiles = N >> 4;
  for (int tile = blockIdx.x; tile < ntiles; tile += gridDim.x) {
    // stage A: float4 load -> packed 4xbf16 store (b64)
    const float4* xr = (const float4*)(xn + (size_t)tile * 16 * D);
    for (int i = tid; i < 16 * D / 4; i += 256) {
      float4 vv = xr[i];
      ((uint2*)As)[i] = make_uint2(bfbits(vv.x) | (bfbits(vv.y) << 16),
                                   bfbits(vv.z) | (bfbits(vv.w) << 16));
    }
    __syncthreads();
    // prefetch all A fragments, then run the WMMA chain
    v16bf a0 = load_A_frag(As, lane, 0);
    v16bf a1 = load_A_frag(As, lane, 32);
    v16bf a2 = load_A_frag(As, lane, 64);
    v16bf a3 = load_A_frag(As, lane, 96);
    v8f acc = {};
    acc = wmma_bf16(a0, bf[0], acc);
    acc = wmma_bf16(a1, bf[1], acc);
    acc = wmma_bf16(a2, bf[2], acc);
    acc = wmma_bf16(a3, bf[3], acc);
#pragma unroll
    for (int r = 0; r < 8; ++r) {
      int m = r + ((lane & 16) ? 8 : 0);
      q[((size_t)tile * 16 + m) * D + n] = acc[r] + bias;
    }
    __syncthreads();
  }
}

// Fused K+V edge GEMM. One A-tile (xn[src]+te) feeds both Wk and Wv WMMAs.
__global__ void k_edge_kv(const float* __restrict__ xn, const long long* __restrict__ ei,
                          const float* __restrict__ ewt, const float* __restrict__ et,
                          const float* __restrict__ Wk, const float* __restrict__ bk,
                          const float* __restrict__ Wv, const float* __restrict__ bv,
                          const float* __restrict__ q, const float* __restrict__ deg,
                          float* __restrict__ attn, __bf16* __restrict__ vws, int E) {
  __shared__ __bf16 As[16 * D];
  __shared__ float  divs[D];       // sinusoidal div_term table
  __shared__ long long sSrc[16];
  __shared__ float  sT[16];
  int tid = threadIdx.x, lane = tid & 31, w = tid >> 5;   // wave = head
  if (tid < D) divs[tid] = __powf(10000.f, -(float)(2 * (tid >> 1)) * (1.0f / 128.0f));
  v16bf bkf[4], bvf[4];
#pragma unroll
  for (int ks = 0; ks < 4; ++ks) {
    bkf[ks] = load_B_frag_global(Wk, lane, ks * 32, w);
    bvf[ks] = load_B_frag_global(Wv, lane, ks * 32, w);
  }
  int dk = lane & 15, hi = (lane >> 4) & 1;
  float biasK = bk[w * DKK + dk];
  float biasV = bv[w * DKK + dk];
  __syncthreads();
  int etiles = E >> 4;
  for (int tile = blockIdx.x; tile < etiles; tile += gridDim.x) {
    if (tid < 16) {
      sSrc[tid] = ei[tile * 16 + tid];
      sT[tid]   = et[tile * 16 + tid];
    }
    __syncthreads();
    // stage A = xn[src] + temporal_encoding : float4 gather -> b64 packed store
    for (int i = tid; i < 16 * D / 4; i += 256) {
      int el = i >> 5;
      int d0 = (i & 31) * 4;
      float4 xv = *(const float4*)(xn + (size_t)sSrc[el] * D + d0);
      float t = sT[el] * 200.f;
      float e0 = xv.x + __sinf(t * divs[d0]);
      float e1 = xv.y + __cosf(t * divs[d0 + 1]);
      float e2 = xv.z + __sinf(t * divs[d0 + 2]);
      float e3 = xv.w + __cosf(t * divs[d0 + 3]);
      ((uint2*)As)[i] = make_uint2(bfbits(e0) | (bfbits(e1) << 16),
                                   bfbits(e2) | (bfbits(e3) << 16));
    }
    __syncthreads();
    v16bf a0 = load_A_frag(As, lane, 0);
    v16bf a1 = load_A_frag(As, lane, 32);
    v16bf a2 = load_A_frag(As, lane, 64);
    v16bf a3 = load_A_frag(As, lane, 96);
    v8f accK = {}, accV = {};
    accK = wmma_bf16(a0, bkf[0], accK);
    accV = wmma_bf16(a0, bvf[0], accV);
    accK = wmma_bf16(a1, bkf[1], accK);
    accV = wmma_bf16(a1, bvf[1], accV);
    accK = wmma_bf16(a2, bkf[2], accK);
    accV = wmma_bf16(a2, bvf[2], accV);
    accK = wmma_bf16(a3, bkf[3], accK);
    accV = wmma_bf16(a3, bvf[3], accV);
    // ---- K path: attn[e,h] = <k, q[dst,h]> * 0.25 * (w[e]/deg[src]) ----
#pragma unroll
    for (int r = 0; r < 8; ++r) {
      int eloc = r + 8 * hi;
      int e    = tile * 16 + eloc;
      long long dst = ei[E + e];
      float p = (accK[r] + biasK) * q[(size_t)dst * D + w * DKK + dk];
#pragma unroll
      for (int m = 8; m >= 1; m >>= 1) p += __shfl_xor(p, m, 32);  // reduce over dk
      if (dk == 0) {
        float dg = deg[sSrc[eloc]];
        float sc = (dg > 0.f) ? (ewt[e] / dg) : 0.f;
        attn[(size_t)e * H + w] = p * 0.25f * sc;
      }
    }
    // ---- V path: store bf16 v to workspace ----
#pragma unroll
    for (int r = 0; r < 8; ++r) {
      int e = tile * 16 + r + 8 * hi;
      vws[(size_t)e * D + w * DKK + dk] = f2bf(accV[r] + biasV);
    }
    __syncthreads();
  }
}

// scatter-max via monotonic uint key
__global__ void k_attn_max(const long long* __restrict__ ei, const float* __restrict__ attn,
                           unsigned* __restrict__ maxu, int E) {
  int i = blockIdx.x * blockDim.x + threadIdx.x;
  if (i >= E * H) return;
  int e = i >> 3, h = i & 7;
  long long d = ei[E + e];
  unsigned b = __float_as_uint(attn[i]);
  unsigned key = (b & 0x80000000u) ? ~b : (b | 0x80000000u);
  atomicMax(&maxu[(size_t)d * H + h], key);
}

__global__ void k_attn_exp(const long long* __restrict__ ei, float* __restrict__ attn,
                           const unsigned* __restrict__ maxu, float* __restrict__ den, int E) {
  int i = blockIdx.x * blockDim.x + threadIdx.x;
  if (i >= E * H) return;
  int e = i >> 3, h = i & 7;
  long long d = ei[E + e];
  unsigned key = maxu[(size_t)d * H + h];
  float m = (key & 0x80000000u) ? __uint_as_float(key ^ 0x80000000u) : __uint_as_float(~key);
  float ex = __expf(attn[i] - m);
  attn[i] = ex;
  atomicAdd(&den[(size_t)d * H + h], ex);
}

// one wave per edge: 128-wide attn*v scatter-add (L2-resident target)
__global__ void k_aggregate(const long long* __restrict__ ei, const float* __restrict__ attn,
                            const float* __restrict__ den, const __bf16* __restrict__ vws,
                            float* __restrict__ out, int E, int nwaves) {
  int gw   = (blockIdx.x * blockDim.x + threadIdx.x) >> 5;
  int lane = threadIdx.x & 31;
  for (int e = gw; e < E; e += nwaves) {
    long long d = ei[E + e];
    int h = lane >> 2;
    float coef = attn[(size_t)e * H + h] / den[(size_t)d * H + h];
    int c0 = lane * 4;
    const __bf16* vp = vws + (size_t)e * D + c0;
#pragma unroll
    for (int j = 0; j < 4; ++j)
      atomicAdd(&out[(size_t)d * D + c0 + j], coef * bf2f(vp[j]));
  }
}

__global__ void k_final(const float* __restrict__ x, float* __restrict__ out, int N) {
  int i = blockIdx.x * blockDim.x + threadIdx.x;
  if (i >= N * D) return;
  float a = out[i];
  out[i] = x[i] + 0.5f * a * (1.f + erff(a * 0.70710678118654752f));
}

// ---- launcher --------------------------------------------------------------
extern "C" void kernel_launch(void* const* d_in, const int* in_sizes, int n_in,
                              void* d_out, int out_size, void* d_ws, size_t ws_size,
                              hipStream_t stream) {
  const float*     x   = (const float*)d_in[0];
  const long long* ei  = (const long long*)d_in[1];
  const float*     ewt = (const float*)d_in[2];
  const float*     et  = (const float*)d_in[4];
  const float*     Wk  = (const float*)d_in[5];
  const float*     bk  = (const float*)d_in[6];
  const float*     Wq  = (const float*)d_in[7];
  const float*     bq  = (const float*)d_in[8];
  const float*     Wv  = (const float*)d_in[9];
  const float*     bv  = (const float*)d_in[10];
  int N = in_sizes[3];   // x_time length
  int E = in_sizes[2];   // edge_weight length
  float* out = (float*)d_out;

  char* ws = (char*)d_ws;
  size_t off = 0;
  auto take = [&](size_t bytes) { void* p = ws + off; off += (bytes + 255) & ~(size_t)255; return p; };
  float*    xn   = (float*)   take((size_t)N * D * 4);
  float*    qn   = (float*)   take((size_t)N * D * 4);
  float*    deg  = (float*)   take((size_t)N * 4);
  unsigned* maxu = (unsigned*)take((size_t)N * H * 4);
  float*    den  = (float*)   take((size_t)N * H * 4);
  float*    attn = (float*)   take((size_t)E * H * 4);
  __bf16*   vws  = (__bf16*)  take((size_t)E * D * 2);

  k_init<<<(N * D + 255) / 256, 256, 0, stream>>>(out, deg, maxu, den, N);
  k_layernorm<<<(N + 7) / 8, 256, 0, stream>>>(x, xn, N);
  k_deg<<<(E + 255) / 256, 256, 0, stream>>>(ei, ewt, deg, E);
  k_proj_q<<<1024, 256, 0, stream>>>(xn, Wq, bq, qn, N);
  k_edge_kv<<<2048, 256, 0, stream>>>(xn, ei, ewt, et, Wk, bk, Wv, bv, qn, deg, attn, vws, E);
  k_attn_max<<<(E * H + 255) / 256, 256, 0, stream>>>(ei, attn, maxu, E);
  k_attn_exp<<<(E * H + 255) / 256, 256, 0, stream>>>(ei, attn, maxu, den, E);
  {
    int blocks = 2048;
    int nwaves = blocks * 256 / 32;
    k_aggregate<<<blocks, 256, 0, stream>>>(ei, attn, den, vws, out, E, nwaves);
  }
  k_final<<<(N * D + 255) / 256, 256, 0, stream>>>(x, out, N);
}
